// GNNTransformer_16320875725529
// MI455X (gfx1250) — matrix operand order repaired and checked by reading, not compiled, not run on hardware
//
#include <hip/hip_runtime.h>
#include <hip/hip_bf16.h>

// ---------------------------------------------------------------------------
// GNN (CGCNN conv x3) + Transformer encoder x5 forward, fp32 in/out.
// GEMMs: v_wmma_f32_16x16x32_f16, A tile staged global->LDS via
// global_load_async_to_lds_b128 (ASYNCcnt) and shared by 4 waves/block.
// Conv GEMM decomposed through the concat: gated = xW1[n] + xW2[idx] + nbrW3 + b
// Intermediate conv products stored fp16 (BN passes are bandwidth-bound).
// ---------------------------------------------------------------------------

typedef __attribute__((ext_vector_type(16))) _Float16 v16h;
typedef __attribute__((ext_vector_type(8)))  _Float16 v8h;
typedef __attribute__((ext_vector_type(8)))  float    v8f;

#define CB 1024          // batch (crystals)
#define APC 48           // atoms per crystal
#define NATOM (CB*APC)   // 49152
#define MNBR 12
#define NMROWS (NATOM*MNBR) // 589824
#define EPS 1e-5f

__device__ __forceinline__ float sp_f(float x) {
    return (x > 20.f) ? x : log1pf(expf(x));
}
__device__ __forceinline__ float sig_f(float x) {
    return 1.f / (1.f + expf(-x));
}

// ---------------------------------------------------------------------------
// WMMA GEMM: C[M,N] = act( resid + A[M,Kp] @ BT[N,Kp]^T + bias )
// A, BT fp16 row-major, Kp multiple of 32 (<=256), M multiple of 16.
// 4 waves / block; block stages its 16xKp A tile (contiguous 32*Kp bytes)
// into LDS with async global->LDS b128 copies, then all 4 waves read A
// fragments from LDS while streaming B fragments from global.
// act: 0=none 1=softplus 2=relu
// ---------------------------------------------------------------------------
__global__ void gemm_wmma(const _Float16* __restrict__ A,
                          const _Float16* __restrict__ BT,
                          const float* __restrict__ bias,
                          const float* __restrict__ resid,
                          float* __restrict__ C,
                          _Float16* __restrict__ Ch,
                          int Mr, int Nc, int Kp, int act) {
    __shared__ _Float16 shA[16 * 256];   // 8 KB, Kp <= 256

    const int tid  = threadIdx.x;
    const int wave = tid >> 5;
    const int lane = tid & 31;
    const int tn = blockIdx.x * 4 + wave;
    const int tm = blockIdx.y;

    // ---- stage A tile: rows [tm*16, tm*16+16) x Kp halves, contiguous ----
    {
        const _Float16* Abase = A + (size_t)tm * 16 * Kp;
        const unsigned lds0 = (unsigned)(size_t)(&shA[0]);
        const int chunks = 2 * Kp;            // 16-byte chunks in tile
        for (int c = tid; c < chunks; c += 128) {
            unsigned goff = (unsigned)(c * 16);
            unsigned lds  = lds0 + (unsigned)(c * 16);
            asm volatile("global_load_async_to_lds_b128 %0, %1, %2"
                         :: "v"(lds), "v"(goff), "s"(Abase) : "memory");
        }
        asm volatile("s_wait_asynccnt 0x0" ::: "memory");
    }
    __syncthreads();

    if (tn * 16 < Nc) {
        const int l16 = lane & 15;
        const int col = tn * 16 + l16;          // B column striped across lanes
        const int akb = (lane < 16) ? 0 : 8;    // A frag K split (ISA 16-bit A)
        const int bkb = (lane < 16) ? 0 : 8;    // B frag K split (same as A)

        v8f acc;
#pragma unroll
        for (int i = 0; i < 8; ++i) acc[i] = 0.f;

        const _Float16* arow = shA + (size_t)l16 * Kp;   // LDS
        const _Float16* brow = BT + (size_t)col * Kp;    // global

        for (int k0 = 0; k0 < Kp; k0 += 32) {
            const _Float16* ap = arow + k0 + akb;
            v8h alo = *(const v8h*)(ap);
            v8h ahi = *(const v8h*)(ap + 16);
            v16h av, bv;
            v8h blo = *(const v8h*)(brow + k0 + bkb);
            v8h bhi = *(const v8h*)(brow + k0 + bkb + 16);
#pragma unroll
            for (int i = 0; i < 8; ++i) { av[i] = alo[i]; av[i + 8] = ahi[i]; bv[i] = blo[i]; bv[i + 8] = bhi[i]; }
            acc = __builtin_amdgcn_wmma_f32_16x16x32_f16(
                      false, av, false, bv, (short)0, acc, false, false);
        }

        const int nc = tn * 16 + l16;
        const float bv_ = bias ? bias[nc] : 0.f;
#pragma unroll
        for (int v = 0; v < 8; ++v) {
            int mr = tm * 16 + v + ((lane >> 4) << 3);  // D layout: VGPR v -> M
            float val = acc[v] + bv_;
            size_t o = (size_t)mr * Nc + nc;
            if (resid) val += resid[o];
            if (act == 1) val = sp_f(val);
            else if (act == 2) val = fmaxf(val, 0.f);
            if (C)  C[o]  = val;
            if (Ch) Ch[o] = (_Float16)val;
        }
    }
}

// --- fp32 [R,K] -> fp16 [R,Kp] (zero-padded K) ------------------------------
__global__ void padcvt(const float* __restrict__ src, _Float16* __restrict__ dst,
                       int R, int K, int Kp) {
    size_t i = (size_t)blockIdx.x * blockDim.x + threadIdx.x;
    size_t tot = (size_t)R * Kp;
    if (i >= tot) return;
    int k = (int)(i % Kp);
    int r = (int)(i / Kp);
    dst[i] = (k < K) ? (_Float16)src[(size_t)r * K + k] : (_Float16)0.f;
}

// --- weight fp32 [K,N] (row stride ldn) -> fp16 transposed [N,Kp] -----------
__global__ void wtcvt(const float* __restrict__ src, _Float16* __restrict__ dst,
                      int K, int N, int Kp, int ldn) {
    size_t i = (size_t)blockIdx.x * blockDim.x + threadIdx.x;
    size_t tot = (size_t)N * Kp;
    if (i >= tot) return;
    int k = (int)(i % Kp);
    int n = (int)(i / Kp);
    dst[i] = (k < K) ? (_Float16)src[(size_t)k * ldn + n] : (_Float16)0.f;
}

__global__ void f2h(const float* __restrict__ src, _Float16* __restrict__ dst, size_t n) {
    size_t i = (size_t)blockIdx.x * blockDim.x + threadIdx.x;
    if (i < n) dst[i] = (_Float16)src[i];
}

__global__ void zero_f(float* p, int n) {
    int i = blockIdx.x * blockDim.x + threadIdx.x;
    if (i < n) p[i] = 0.f;
}

// --- BN1 stats over gated rows (recompute gather, no materialization) -------
// stats layout: [0,C)=sum [C,2C)=sumsq [2C,3C)=mean [3C,4C)=invstd
__global__ void conv_stats(const _Float16* __restrict__ xW1,
                           const _Float16* __restrict__ xW2,
                           const _Float16* __restrict__ nW3,
                           const int* __restrict__ idx,
                           const float* __restrict__ cb, float* __restrict__ stats,
                           int rowsPerBlk) {
    int c = threadIdx.x;          // 128 channels
    int r0 = blockIdx.x * rowsPerBlk;
    float s = 0.f, s2 = 0.f;
    for (int r = r0; r < r0 + rowsPerBlk; ++r) {
        int n = r / MNBR;
        int j = idx[r];
        float g = (float)xW1[(size_t)n * 128 + c] + (float)xW2[(size_t)j * 128 + c]
                + (float)nW3[(size_t)r * 128 + c] + cb[c];
        s += g; s2 += g * g;
    }
    atomicAdd(&stats[c], s);
    atomicAdd(&stats[128 + c], s2);
}

__global__ void bn_finalize(float* stats, int C, float cnt) {
    int c = threadIdx.x;
    if (c >= C) return;
    float mean = stats[c] / cnt;
    float var  = stats[C + c] / cnt - mean * mean;
    stats[2 * C + c] = mean;
    stats[3 * C + c] = rsqrtf(var + EPS);
}

// --- BN1 apply + sigmoid*softplus + sum over neighbors ----------------------
__global__ void conv_apply(const _Float16* __restrict__ xW1,
                           const _Float16* __restrict__ xW2,
                           const _Float16* __restrict__ nW3,
                           const int* __restrict__ idx,
                           const float* __restrict__ cb, const float* __restrict__ stats,
                           const float* __restrict__ g1, const float* __restrict__ b1,
                           float* __restrict__ summed) {
    int n = blockIdx.x;
    int c = threadIdx.x;  // 128
    __shared__ float gg[128];
    float mean = stats[256 + c], inv = stats[384 + c];
    float w1v = (float)xW1[(size_t)n * 128 + c];
    float acc = 0.f;
    for (int m = 0; m < MNBR; ++m) {
        int r = n * MNBR + m;
        int j = idx[r];
        float g = w1v + (float)xW2[(size_t)j * 128 + c]
                + (float)nW3[(size_t)r * 128 + c] + cb[c];
        gg[c] = g1[c] * (g - mean) * inv + b1[c];
        __syncthreads();
        if (c < 64) acc += sig_f(gg[c]) * sp_f(gg[c + 64]);
        __syncthreads();
    }
    if (c < 64) summed[(size_t)n * 64 + c] = acc;
}

// --- generic row stats over X[R,C] (C==blockDim.x) --------------------------
__global__ void row_stats(const float* __restrict__ X, float* __restrict__ stats,
                          int rowsPerBlk, int C) {
    int c = threadIdx.x;
    int r0 = blockIdx.x * rowsPerBlk;
    float s = 0.f, s2 = 0.f;
    for (int r = r0; r < r0 + rowsPerBlk; ++r) {
        float v = X[(size_t)r * C + c];
        s += v; s2 += v * v;
    }
    atomicAdd(&stats[c], s);
    atomicAdd(&stats[C + c], s2);
}

// --- x = softplus(x + bn2(summed)) ------------------------------------------
__global__ void conv_update(float* __restrict__ x, const float* __restrict__ summed,
                            const float* __restrict__ stats,
                            const float* __restrict__ g2, const float* __restrict__ b2) {
    size_t i = (size_t)blockIdx.x * blockDim.x + threadIdx.x;
    if (i >= (size_t)NATOM * 64) return;
    int c = (int)(i & 63);
    float v = g2[c] * (summed[i] - stats[128 + c]) * stats[192 + c] + b2[c];
    x[i] = sp_f(x[i] + v);
}

// --- segment mean (contiguous runs of APC) + softplus -> fp16 ---------------
__global__ void seg_mean(const float* __restrict__ x, _Float16* __restrict__ crysh) {
    int b = blockIdx.x;
    int c = threadIdx.x;  // 64
    float s = 0.f;
    for (int a = 0; a < APC; ++a) s += x[(size_t)(b * APC + a) * 64 + c];
    crysh[(size_t)b * 64 + c] = (_Float16)sp_f(s / (float)APC);
}

// --- assemble (B,8,256) tokens ----------------------------------------------
__global__ void assemble(const float* __restrict__ spg_emb, const int* __restrict__ spg_id,
                         const float* __restrict__ mac, const float* __restrict__ ele,
                         const float* __restrict__ crys, float* __restrict__ total) {
    int bt = blockIdx.x;      // b*8 + t
    int c = threadIdx.x;      // 256
    int b = bt >> 3, t = bt & 7;
    float v;
    if (t < 5)       v = spg_emb[(size_t)spg_id[b * 5 + t] * 256 + c];
    else if (t == 5) v = mac[(size_t)b * 256 + c];
    else if (t == 6) v = ele[(size_t)b * 256 + c];
    else             v = crys[(size_t)b * 256 + c];
    total[(size_t)bt * 256 + c] = v;
}

// --- LayerNorm over C (blockDim.x == C), fp16 and/or fp32 out ---------------
__global__ void ln_kernel(const float* __restrict__ X, const float* __restrict__ g,
                          const float* __restrict__ b, _Float16* __restrict__ outH,
                          float* __restrict__ outF, int C) {
    int row = blockIdx.x, c = threadIdx.x;
    __shared__ float red[256];
    float v = X[(size_t)row * C + c];
    red[c] = v; __syncthreads();
    for (int s = C >> 1; s > 0; s >>= 1) {
        if (c < s) red[c] += red[c + s];
        __syncthreads();
    }
    float mean = red[0] / C; __syncthreads();
    float d = v - mean;
    red[c] = d * d; __syncthreads();
    for (int s = C >> 1; s > 0; s >>= 1) {
        if (c < s) red[c] += red[c + s];
        __syncthreads();
    }
    float inv = rsqrtf(red[0] / C + EPS);
    float y = g[c] * d * inv + b[c];
    size_t o = (size_t)row * C + c;
    if (outH) outH[o] = (_Float16)y;
    if (outF) outF[o] = y;
}

// --- tiny attention: seq=8, H=4, hd=64; one thread per (b,h,q) --------------
__global__ void attention(const float* __restrict__ qkv, _Float16* __restrict__ oh) {
    int tid = blockIdx.x * blockDim.x + threadIdx.x;
    if (tid >= CB * 4 * 8) return;
    int q = tid & 7, h = (tid >> 3) & 3, b = tid >> 5;
    const float* base = qkv + (size_t)b * 8 * 768;
    const float* qp = base + (size_t)q * 768 + h * 64;
    float s[8], mx = -1e30f;
    for (int k = 0; k < 8; ++k) {
        const float* kp = base + (size_t)k * 768 + 256 + h * 64;
        float d = 0.f;
        for (int e = 0; e < 64; ++e) d += qp[e] * kp[e];
        s[k] = d * 0.125f;
        mx = fmaxf(mx, s[k]);
    }
    float den = 0.f;
    for (int k = 0; k < 8; ++k) { s[k] = expf(s[k] - mx); den += s[k]; }
    float rden = 1.f / den;
    _Float16* op = oh + (size_t)(b * 8 + q) * 256 + h * 64;
    for (int e = 0; e < 64; ++e) {
        float a = 0.f;
        for (int k = 0; k < 8; ++k)
            a += s[k] * base[(size_t)k * 768 + 512 + h * 64 + e];
        op[e] = (_Float16)(a * rden);
    }
}

// ---------------------------------------------------------------------------
extern "C" void kernel_launch(void* const* d_in, const int* in_sizes, int n_in,
                              void* d_out, int out_size, void* d_ws, size_t ws_size,
                              hipStream_t stream) {
    (void)in_sizes; (void)n_in; (void)out_size; (void)ws_size;

    const float* ele_vector = (const float*)d_in[0];
    const float* mac_prop   = (const float*)d_in[1];
    const float* atom_fea   = (const float*)d_in[2];
    const float* nbr_fea    = (const float*)d_in[3];
    const float* fc1_w      = (const float*)d_in[4];
    const float* fc1_b      = (const float*)d_in[5];
    const float* spg_emb    = (const float*)d_in[6];
    const float* fc2_w      = (const float*)d_in[7];
    const float* fc2_b      = (const float*)d_in[8];
    const float* emb_w      = (const float*)d_in[9];
    const float* emb_b      = (const float*)d_in[10];
    const float* conv_fc_w  = (const float*)d_in[11];
    const float* conv_fc_b  = (const float*)d_in[12];
    const float* conv_bn1_g = (const float*)d_in[13];
    const float* conv_bn1_b = (const float*)d_in[14];
    const float* conv_bn2_g = (const float*)d_in[15];
    const float* conv_bn2_b = (const float*)d_in[16];
    const float* c2f_w      = (const float*)d_in[17];
    const float* c2f_b      = (const float*)d_in[18];
    const float* ln1_g      = (const float*)d_in[19];
    const float* ln1_b      = (const float*)d_in[20];
    const float* qkv_w      = (const float*)d_in[21];
    const float* qkv_b      = (const float*)d_in[22];
    const float* outp_w     = (const float*)d_in[23];
    const float* outp_b     = (const float*)d_in[24];
    const float* ln2_g      = (const float*)d_in[25];
    const float* ln2_b      = (const float*)d_in[26];
    const float* ff1_w      = (const float*)d_in[27];
    const float* ff1_b      = (const float*)d_in[28];
    const float* ff2_w      = (const float*)d_in[29];
    const float* ff2_b      = (const float*)d_in[30];
    const float* fin_g      = (const float*)d_in[31];
    const float* fin_b      = (const float*)d_in[32];
    const int*   spg_id     = (const int*)d_in[33];
    const int*   nbr_idx    = (const int*)d_in[34];

    // workspace bump allocator (256B aligned)
    char* wbase = (char*)d_ws;
    size_t off = 0;
    auto alloc = [&](size_t bytes) -> void* {
        off = (off + 255) & ~(size_t)255;
        void* p = wbase + off;
        off += bytes;
        return p;
    };

    _Float16* nbrh  = (_Float16*)alloc((size_t)NMROWS * 64 * 2);
    _Float16* atomh = (_Float16*)alloc((size_t)NATOM * 96 * 2);
    _Float16* eleh  = (_Float16*)alloc((size_t)CB * 96 * 2);
    _Float16* mach  = (_Float16*)alloc((size_t)CB * 32 * 2);
    float* x        = (float*)alloc((size_t)NATOM * 64 * 4);
    _Float16* xh    = (_Float16*)alloc((size_t)NATOM * 64 * 2);
    _Float16* xW1h  = (_Float16*)alloc((size_t)NATOM * 128 * 2);
    _Float16* xW2h  = (_Float16*)alloc((size_t)NATOM * 128 * 2);
    _Float16* nW3h  = (_Float16*)alloc((size_t)NMROWS * 128 * 2);
    float* summed   = (float*)alloc((size_t)NATOM * 64 * 4);
    float* bn1s     = (float*)alloc(512 * 4);
    float* bn2s     = (float*)alloc(256 * 4);
    float* ele_out  = (float*)alloc((size_t)CB * 256 * 4);
    float* mac_out  = (float*)alloc((size_t)CB * 256 * 4);
    _Float16* crysh = (_Float16*)alloc((size_t)CB * 64 * 2);
    float* crys_out = (float*)alloc((size_t)CB * 256 * 4);
    float* total    = (float*)alloc((size_t)CB * 8 * 256 * 4);
    _Float16* hh    = (_Float16*)alloc((size_t)CB * 8 * 256 * 2);
    float* qkv      = (float*)alloc((size_t)CB * 8 * 768 * 4);
    _Float16* oh    = (_Float16*)alloc((size_t)CB * 8 * 256 * 2);
    _Float16* g1h   = (_Float16*)alloc((size_t)CB * 8 * 256 * 2);
    _Float16* fc1T  = (_Float16*)alloc((size_t)256 * 96 * 2);
    _Float16* fc2T  = (_Float16*)alloc((size_t)256 * 32 * 2);
    _Float16* embT  = (_Float16*)alloc((size_t)64 * 96 * 2);
    _Float16* c2fT  = (_Float16*)alloc((size_t)256 * 64 * 2);
    _Float16* w1T   = (_Float16*)alloc((size_t)128 * 64 * 2);
    _Float16* w2T   = (_Float16*)alloc((size_t)128 * 64 * 2);
    _Float16* w3T   = (_Float16*)alloc((size_t)128 * 64 * 2);
    _Float16* qkvT  = (_Float16*)alloc((size_t)768 * 256 * 2);
    _Float16* owT   = (_Float16*)alloc((size_t)256 * 256 * 2);
    _Float16* f1T   = (_Float16*)alloc((size_t)256 * 256 * 2);
    _Float16* f2T   = (_Float16*)alloc((size_t)256 * 256 * 2);

    auto launch_padcvt = [&](const float* s, _Float16* d, int R, int K, int Kp) {
        size_t tot = (size_t)R * Kp;
        padcvt<<<dim3((unsigned)((tot + 255) / 256)), dim3(256), 0, stream>>>(s, d, R, K, Kp);
    };
    auto launch_wtcvt = [&](const float* s, _Float16* d, int K, int N, int Kp, int ldn) {
        size_t tot = (size_t)N * Kp;
        wtcvt<<<dim3((unsigned)((tot + 255) / 256)), dim3(256), 0, stream>>>(s, d, K, N, Kp, ldn);
    };
    auto gemm = [&](const _Float16* A, const _Float16* BT, const float* bias,
                    const float* resid, float* C, _Float16* Ch,
                    int Mr, int Nc, int Kp, int act) {
        dim3 g((unsigned)((Nc + 63) / 64), (unsigned)(Mr / 16));
        gemm_wmma<<<g, dim3(128), 0, stream>>>(A, BT, bias, resid, C, Ch, Mr, Nc, Kp, act);
    };

    // ---- input / weight conversions ----
    launch_padcvt(nbr_fea, nbrh, NMROWS, 41, 64);
    launch_padcvt(atom_fea, atomh, NATOM, 92, 96);
    launch_padcvt(ele_vector, eleh, CB, 92, 96);
    launch_padcvt(mac_prop, mach, CB, 14, 32);
    launch_wtcvt(fc1_w, fc1T, 92, 256, 96, 256);
    launch_wtcvt(fc2_w, fc2T, 14, 256, 32, 256);
    launch_wtcvt(emb_w, embT, 92, 64, 96, 64);
    launch_wtcvt(c2f_w, c2fT, 64, 256, 64, 256);

    // ---- front GEMMs ----
    gemm(eleh, fc1T, fc1_b, nullptr, ele_out, nullptr, CB, 256, 96, 1);
    gemm(mach, fc2T, fc2_b, nullptr, mac_out, nullptr, CB, 256, 32, 1);
    gemm(atomh, embT, emb_b, nullptr, x, nullptr, NATOM, 64, 96, 0);

    // ---- 3 conv layers ----
    for (int l = 0; l < 3; ++l) {
        const float* wl = conv_fc_w + (size_t)l * 169 * 128;
        const float* cb = conv_fc_b + (size_t)l * 128;
        launch_wtcvt(wl,             w1T, 64, 128, 64, 128);
        launch_wtcvt(wl + 64 * 128,  w2T, 64, 128, 64, 128);
        launch_wtcvt(wl + 128 * 128, w3T, 41, 128, 64, 128);

        f2h<<<dim3((NATOM * 64 + 255) / 256), dim3(256), 0, stream>>>(x, xh, (size_t)NATOM * 64);
        gemm(xh, w1T, nullptr, nullptr, nullptr, xW1h, NATOM, 128, 64, 0);
        gemm(xh, w2T, nullptr, nullptr, nullptr, xW2h, NATOM, 128, 64, 0);
        gemm(nbrh, w3T, nullptr, nullptr, nullptr, nW3h, NMROWS, 128, 64, 0);

        zero_f<<<dim3(2), dim3(256), 0, stream>>>(bn1s, 512);
        conv_stats<<<dim3(NMROWS / 64), dim3(128), 0, stream>>>(
            xW1h, xW2h, nW3h, nbr_idx, cb, bn1s, 64);
        bn_finalize<<<dim3(1), dim3(128), 0, stream>>>(bn1s, 128, (float)NMROWS);
        conv_apply<<<dim3(NATOM), dim3(128), 0, stream>>>(
            xW1h, xW2h, nW3h, nbr_idx, cb, bn1s,
            conv_bn1_g + (size_t)l * 128, conv_bn1_b + (size_t)l * 128, summed);

        zero_f<<<dim3(1), dim3(256), 0, stream>>>(bn2s, 256);
        row_stats<<<dim3(NATOM / 64), dim3(64), 0, stream>>>(summed, bn2s, 64, 64);
        bn_finalize<<<dim3(1), dim3(64), 0, stream>>>(bn2s, 64, (float)NATOM);
        conv_update<<<dim3((NATOM * 64 + 255) / 256), dim3(256), 0, stream>>>(
            x, summed, bn2s, conv_bn2_g + (size_t)l * 64, conv_bn2_b + (size_t)l * 64);
    }

    // ---- pooling + crystal features ----
    seg_mean<<<dim3(CB), dim3(64), 0, stream>>>(x, crysh);
    gemm(crysh, c2fT, c2f_b, nullptr, crys_out, nullptr, CB, 256, 64, 1);
    assemble<<<dim3(CB * 8), dim3(256), 0, stream>>>(spg_emb, spg_id, mac_out, ele_out,
                                                     crys_out, total);

    // ---- 5 transformer encoder layers ----
    const int S = CB * 8;  // 8192 rows
    for (int l = 0; l < 5; ++l) {
        launch_wtcvt(qkv_w + (size_t)l * 256 * 768, qkvT, 256, 768, 256, 768);
        launch_wtcvt(outp_w + (size_t)l * 256 * 256, owT, 256, 256, 256, 256);
        launch_wtcvt(ff1_w + (size_t)l * 256 * 256, f1T, 256, 256, 256, 256);
        launch_wtcvt(ff2_w + (size_t)l * 256 * 256, f2T, 256, 256, 256, 256);

        ln_kernel<<<dim3(S), dim3(256), 0, stream>>>(
            total, ln1_g + (size_t)l * 256, ln1_b + (size_t)l * 256, hh, nullptr, 256);
        gemm(hh, qkvT, qkv_b + (size_t)l * 768, nullptr, qkv, nullptr, S, 768, 256, 0);
        attention<<<dim3((CB * 32 + 63) / 64), dim3(64), 0, stream>>>(qkv, oh);
        gemm(oh, owT, outp_b + (size_t)l * 256, total, total, nullptr, S, 256, 256, 0);

        ln_kernel<<<dim3(S), dim3(256), 0, stream>>>(
            total, ln2_g + (size_t)l * 256, ln2_b + (size_t)l * 256, hh, nullptr, 256);
        gemm(hh, f1T, ff1_b + (size_t)l * 256, nullptr, nullptr, g1h, S, 256, 256, 2);
        gemm(g1h, f2T, ff2_b + (size_t)l * 256, total, total, nullptr, S, 256, 256, 0);
    }

    // ---- final LN straight into d_out ----
    ln_kernel<<<dim3(S), dim3(256), 0, stream>>>(total, fin_g, fin_b, nullptr,
                                                 (float*)d_out, 256);
}
